// MSFEncoderLayer_54735063220798
// MI455X (gfx1250) — matrix operand order repaired and checked by reading, not compiled, not run on hardware
//
#include <hip/hip_runtime.h>
#include <hip/hip_bf16.h>
#include <stdint.h>

// ---------------------------------------------------------------------------
// MSFEncoderLayer for MI455X (gfx1250, wave32, WMMA bf16 16x16x32, f32 accum)
// ---------------------------------------------------------------------------

typedef __attribute__((ext_vector_type(16))) __bf16          v16bf;
typedef __attribute__((ext_vector_type(8)))  float           v8f;
typedef __attribute__((ext_vector_type(8)))  unsigned short  v8u16;
typedef __attribute__((ext_vector_type(4)))  int             v4i;

#ifndef __has_builtin
#define __has_builtin(x) 0
#endif
#if __has_builtin(__builtin_amdgcn_global_load_async_to_lds_b128)
#define HAVE_ASYNC_LDS 1
#else
#define HAVE_ASYNC_LDS 0
#endif

#if HAVE_ASYNC_LDS
__device__ __forceinline__ void wait_async0() {
#if __has_builtin(__builtin_amdgcn_s_wait_asynccnt)
    __builtin_amdgcn_s_wait_asynccnt(0);
#else
    asm volatile("s_wait_asynccnt 0x0" ::: "memory");
#endif
}
#endif

__device__ __forceinline__ unsigned short f2bf(float f) {
    unsigned int u = __float_as_uint(f);
    u += 0x7FFFu + ((u >> 16) & 1u);          // round-to-nearest-even
    return (unsigned short)(u >> 16);
}
__device__ __forceinline__ float bf2f(unsigned short h) {
    return __uint_as_float(((unsigned int)h) << 16);
}

// --------------------------------------------------------------- f32 -> bf16
__global__ void cvt_f32_bf16(const float* __restrict__ in,
                             unsigned short* __restrict__ out, long long n) {
    long long i = (long long)blockIdx.x * blockDim.x + threadIdx.x;
    long long stride = (long long)gridDim.x * blockDim.x;
    for (; i < n; i += stride) out[i] = f2bf(in[i]);
}

// ---------------------------------------------------------------------------
// WMMA GEMM:  out[m,n] = act( A[m,:K] . W[n,:K] + bias[n] ) (+ res[m,n])
// A: bf16 row-major MxK.  W: bf16 row-major NxK (y = x @ W^T).
// Block = 256 thr = 8 waves (2 Mwaves x 4 Nwaves) -> tile 64M x 128N.
// K staged in macro-steps of 64 via async DMA to double-buffered LDS;
// per macro-step each wave issues 8 WMMAs between sync points.
// Requires M%64==0, N%128==0, K%64==0.
// ---------------------------------------------------------------------------
__global__ __launch_bounds__(256)
void gemm_bias(const unsigned short* __restrict__ A,
               const unsigned short* __restrict__ W,
               const float* __restrict__ bias,
               const float* __restrict__ res,
               float* __restrict__ outf,
               unsigned short* __restrict__ outbf,
               int M, int N, int K, int act) {
    __shared__ unsigned short As[2][64 * 72];    // 64 rows, stride 72 (144B, 16B-aligned)
    const int t = threadIdx.x;
    const int lane = t & 31, wave = t >> 5;
    const int wm = wave >> 2;                    // 0..1
    const int wn = wave & 3;                     // 0..3
    const int m0 = blockIdx.y * 64;
    const int n0 = blockIdx.x * 128;

    const int l15 = lane & 15;
    const int lh  = lane >> 4;                   // 0/1
    const int kb  = lh * 8;                      // A-frag K base (ISA layout)
    const int kboff = lh * 8;                    // B-frag K base

    // A staging: 32B (16 bf16) per thread per macro-step (2 async b128)
    const int srow = t >> 2;                     // 0..63
    const int scg  = (t & 3) * 16;               // 0,16,32,48
    const unsigned short* Ag = A + (size_t)(m0 + srow) * K + scg;
    unsigned short* Lbase0 = &As[0][srow * 72 + scg];
    unsigned short* Lbase1 = &As[1][srow * 72 + scg];

    // B fragment columns (two 16-wide N tiles per wave)
    const int ncol0 = n0 + wn * 32 + l15;
    const unsigned short* Wg0 = W + (size_t)ncol0 * K + kboff;
    const unsigned short* Wg1 = W + (size_t)(ncol0 + 16) * K + kboff;

    v8f acc00 = {0.f,0.f,0.f,0.f,0.f,0.f,0.f,0.f};
    v8f acc01 = acc00, acc10 = acc00, acc11 = acc00;

    union frag { v16bf v; v8u16 h[2]; };

    const int msteps = K >> 6;

#if HAVE_ASYNC_LDS
    // prologue: stage macro-step 0 into buffer 0
    __builtin_amdgcn_global_load_async_to_lds_b128((v4i*)Ag, (v4i*)Lbase0, 0, 0);
    __builtin_amdgcn_global_load_async_to_lds_b128((v4i*)(Ag + 8), (v4i*)(Lbase0 + 8), 0, 0);
#endif

    for (int i = 0; i < msteps; ++i) {
        const int cur = i & 1;
        const int kk = i << 6;
#if HAVE_ASYNC_LDS
        wait_async0();
        __syncthreads();
        if (i + 1 < msteps) {                    // stage next macro-step (other buf)
            const unsigned short* G = Ag + kk + 64;
            unsigned short* L = cur ? Lbase0 : Lbase1;
            __builtin_amdgcn_global_load_async_to_lds_b128((v4i*)G, (v4i*)L, 0, 0);
            __builtin_amdgcn_global_load_async_to_lds_b128((v4i*)(G + 8), (v4i*)(L + 8), 0, 0);
        }
#else
        __syncthreads();
        {
            unsigned short* L = cur ? Lbase1 : Lbase0;
            const unsigned short* G = Ag + kk;
            *(v8u16*)L       = *(const v8u16*)G;
            *(v8u16*)(L + 8) = *(const v8u16*)(G + 8);
        }
        __syncthreads();
#endif
        // ---- B fragments for this macro-step: 2 n-tiles x 2 k-sub-steps
        frag b00, b01, b10, b11;                 // b<ks><nt>
        b00.h[0] = *(const v8u16*)(Wg0 + kk);
        b00.h[1] = *(const v8u16*)(Wg0 + kk + 16);
        b01.h[0] = *(const v8u16*)(Wg1 + kk);
        b01.h[1] = *(const v8u16*)(Wg1 + kk + 16);
        b10.h[0] = *(const v8u16*)(Wg0 + kk + 32);
        b10.h[1] = *(const v8u16*)(Wg0 + kk + 48);
        b11.h[0] = *(const v8u16*)(Wg1 + kk + 32);
        b11.h[1] = *(const v8u16*)(Wg1 + kk + 48);

        const unsigned short* Ab = As[cur];
        const int r0 = (wm * 32 + l15) * 72;
        const int r1 = (wm * 32 + 16 + l15) * 72;

        frag a0, a1;
        // ---- k-sub-step 0 (K = kk..kk+31)
        a0.h[0] = *(const v8u16*)&Ab[r0 + kb];
        a0.h[1] = *(const v8u16*)&Ab[r0 + kb + 16];
        a1.h[0] = *(const v8u16*)&Ab[r1 + kb];
        a1.h[1] = *(const v8u16*)&Ab[r1 + kb + 16];
        acc00 = __builtin_amdgcn_wmma_f32_16x16x32_bf16(false, a0.v, false, b00.v, (short)0, acc00, false, false);
        acc01 = __builtin_amdgcn_wmma_f32_16x16x32_bf16(false, a0.v, false, b01.v, (short)0, acc01, false, false);
        acc10 = __builtin_amdgcn_wmma_f32_16x16x32_bf16(false, a1.v, false, b00.v, (short)0, acc10, false, false);
        acc11 = __builtin_amdgcn_wmma_f32_16x16x32_bf16(false, a1.v, false, b01.v, (short)0, acc11, false, false);
        // ---- k-sub-step 1 (K = kk+32..kk+63)
        a0.h[0] = *(const v8u16*)&Ab[r0 + 32 + kb];
        a0.h[1] = *(const v8u16*)&Ab[r0 + 32 + kb + 16];
        a1.h[0] = *(const v8u16*)&Ab[r1 + 32 + kb];
        a1.h[1] = *(const v8u16*)&Ab[r1 + 32 + kb + 16];
        acc00 = __builtin_amdgcn_wmma_f32_16x16x32_bf16(false, a0.v, false, b10.v, (short)0, acc00, false, false);
        acc01 = __builtin_amdgcn_wmma_f32_16x16x32_bf16(false, a0.v, false, b11.v, (short)0, acc01, false, false);
        acc10 = __builtin_amdgcn_wmma_f32_16x16x32_bf16(false, a1.v, false, b10.v, (short)0, acc10, false, false);
        acc11 = __builtin_amdgcn_wmma_f32_16x16x32_bf16(false, a1.v, false, b11.v, (short)0, acc11, false, false);
    }

    // ---------------- epilogue: 4 tiles of 16x16
    float av[4][8];
#pragma unroll
    for (int r = 0; r < 8; ++r) { av[0][r] = acc00[r]; av[1][r] = acc01[r];
                                  av[2][r] = acc10[r]; av[3][r] = acc11[r]; }
#pragma unroll
    for (int mt = 0; mt < 2; ++mt) {
        const int mbase = m0 + wm * 32 + mt * 16 + lh * 8;
#pragma unroll
        for (int nt = 0; nt < 2; ++nt) {
            const int nc = n0 + wn * 32 + nt * 16 + l15;
            const float bv = bias[nc];
#pragma unroll
            for (int r = 0; r < 8; ++r) {
                size_t idx = (size_t)(mbase + r) * N + nc;
                float v = av[mt * 2 + nt][r] + bv;
                if (act) v = fmaxf(v, 0.f);
                if (res) v += res[idx];
                if (outf)  outf[idx]  = v;
                if (outbf) outbf[idx] = f2bf(v);
            }
        }
    }
}

// ---------------------------------------------------------------------------
// Mix self-attention: per (b,h): q,k,v = (128 x 32) bf16; flash-style softmax.
// grid = 1024*8, block = 256 (8 waves; wave w owns query rows w*16..w*16+15).
// qkv layout: [(l*1024+b)*768 + h*32 + j] (+256 k, +512 v).
// o layout:   [(l*1024+b)*256 + h*32 + j] bf16.
// ---------------------------------------------------------------------------
__global__ __launch_bounds__(256)
void attn_mix(const unsigned short* __restrict__ qkv,
              unsigned short* __restrict__ o) {
    __shared__ unsigned short qs[128 * 40];   // 10240 B
    __shared__ unsigned short ks[128 * 40];   // 10240 B
    __shared__ unsigned short vts[32 * 136];  // v transposed [j][l], 8704 B
    __shared__ unsigned short P[128 * 136];   // probs bf16, 34816 B  (total 64000 B)

    const int b = blockIdx.x >> 3;
    const int h = blockIdx.x & 7;
    const int t = threadIdx.x;
    const int w = t >> 5, lane = t & 31;
    const float SCALE = 0.17677669529663687f;  // 1/sqrt(32)

    // ---- stage q,k,v : each thread loads 16 contiguous bf16 per tensor
    {
        const int l = t >> 1, jg = (t & 1) * 16;
        size_t base = ((size_t)l * 1024 + b) * 768 + h * 32 + jg;
        const v8u16* qp = (const v8u16*)(qkv + base);
        *(v8u16*)&qs[l * 40 + jg]     = qp[0];
        *(v8u16*)&qs[l * 40 + jg + 8] = qp[1];
        const v8u16* kp = (const v8u16*)(qkv + base + 256);
        *(v8u16*)&ks[l * 40 + jg]     = kp[0];
        *(v8u16*)&ks[l * 40 + jg + 8] = kp[1];
        const v8u16* vp = (const v8u16*)(qkv + base + 512);
        union { v8u16 h2[2]; unsigned short u[16]; } vv;
        vv.h2[0] = vp[0]; vv.h2[1] = vp[1];
#pragma unroll
        for (int e = 0; e < 16; ++e) vts[(jg + e) * 136 + l] = vv.u[e];
    }
    __syncthreads();

    // ---- scores: 8 tiles of 16x16, kept in registers (64 f32 / lane)
    union frag { v16bf v; v8u16 h2[2]; };
    frag au, bu;
    const int arow = w * 16 + (lane & 15);
    const int kb   = (lane >> 4) * 8;
    au.h2[0] = *(const v8u16*)&qs[arow * 40 + kb];
    au.h2[1] = *(const v8u16*)&qs[arow * 40 + kb + 16];

    float sc[64];
#pragma unroll
    for (int kt = 0; kt < 8; ++kt) {
        const int krow = kt * 16 + (lane & 15);
        const int koff = (lane >> 4) * 8;
        bu.h2[0] = *(const v8u16*)&ks[krow * 40 + koff];
        bu.h2[1] = *(const v8u16*)&ks[krow * 40 + koff + 16];
        v8f z = {0.f,0.f,0.f,0.f,0.f,0.f,0.f,0.f};
        z = __builtin_amdgcn_wmma_f32_16x16x32_bf16(
                false, au.v, false, bu.v, (short)0, z, false, false);
#pragma unroll
        for (int r = 0; r < 8; ++r) sc[kt * 8 + r] = z[r];
    }

    // ---- row softmax (row r lives on 16 lanes of one half-wave)
#pragma unroll
    for (int r = 0; r < 8; ++r) {
        float m = -1e30f;
#pragma unroll
        for (int kt = 0; kt < 8; ++kt) m = fmaxf(m, sc[kt * 8 + r]);
        m = fmaxf(m, __shfl_xor(m, 8, 32));
        m = fmaxf(m, __shfl_xor(m, 4, 32));
        m = fmaxf(m, __shfl_xor(m, 2, 32));
        m = fmaxf(m, __shfl_xor(m, 1, 32));
        float s = 0.f;
#pragma unroll
        for (int kt = 0; kt < 8; ++kt) {
            float e = __expf((sc[kt * 8 + r] - m) * SCALE);
            sc[kt * 8 + r] = e;
            s += e;
        }
        s += __shfl_xor(s, 8, 32);
        s += __shfl_xor(s, 4, 32);
        s += __shfl_xor(s, 2, 32);
        s += __shfl_xor(s, 1, 32);
        float inv = 1.f / s;
        const int prow = w * 16 + r + ((lane >> 4) << 3);
#pragma unroll
        for (int kt = 0; kt < 8; ++kt)
            P[prow * 136 + kt * 16 + (lane & 15)] = f2bf(sc[kt * 8 + r] * inv);
    }
    __syncthreads();

    // ---- o = P (16x128) @ V (128x32): 2 N-tiles x 4 K-chains
#pragma unroll
    for (int nt = 0; nt < 2; ++nt) {
        v8f oa = {0.f,0.f,0.f,0.f,0.f,0.f,0.f,0.f};
#pragma unroll
        for (int kt2 = 0; kt2 < 4; ++kt2) {
            const int klo = kt2 * 32 + kb;
            au.h2[0] = *(const v8u16*)&P[arow * 136 + klo];
            au.h2[1] = *(const v8u16*)&P[arow * 136 + klo + 16];
            const int n = nt * 16 + (lane & 15);
            const int kr = kt2 * 32 + (lane >> 4) * 8;
            bu.h2[0] = *(const v8u16*)&vts[n * 136 + kr];
            bu.h2[1] = *(const v8u16*)&vts[n * 136 + kr + 16];
            oa = __builtin_amdgcn_wmma_f32_16x16x32_bf16(
                     false, au.v, false, bu.v, (short)0, oa, false, false);
        }
        float ov[8];
#pragma unroll
        for (int r = 0; r < 8; ++r) ov[r] = oa[r];
        const int mbase = w * 16 + ((lane >> 4) << 3);
        const int col = h * 32 + nt * 16 + (lane & 15);
#pragma unroll
        for (int r = 0; r < 8; ++r)
            o[((size_t)(mbase + r) * 1024 + b) * 256 + col] = f2bf(ov[r]);
    }
}

// ---------------------------------------------------------------------------
// Small cross-attention (1 query, 128 keys, 4 heads of 64). grid=1024, blk=256
// ---------------------------------------------------------------------------
__global__ __launch_bounds__(256)
void attn_sa(const unsigned short* __restrict__ qt,
             const unsigned short* __restrict__ kv,
             unsigned short* __restrict__ o) {
    __shared__ float pbuf[4 * 128];
    const int b = blockIdx.x;
    const int t = threadIdx.x;
    const int wave = t >> 5, lane = t & 31;

    if (wave < 4) {
        const int h = wave;
        float sc[4];
#pragma unroll
        for (int rr = 0; rr < 4; ++rr) {
            const int lk = lane * 4 + rr;
            const unsigned short* kr = kv + ((size_t)lk * 1024 + b) * 512 + h * 64;
            const unsigned short* qr = qt + (size_t)b * 256 + h * 64;
            float s = 0.f;
            for (int j = 0; j < 64; ++j) s += bf2f(qr[j]) * bf2f(kr[j]);
            sc[rr] = s * 0.125f;                    // 1/sqrt(64)
        }
        float m = fmaxf(fmaxf(sc[0], sc[1]), fmaxf(sc[2], sc[3]));
        for (int off = 16; off >= 1; off >>= 1) m = fmaxf(m, __shfl_xor(m, off, 32));
        float s = 0.f;
#pragma unroll
        for (int rr = 0; rr < 4; ++rr) { sc[rr] = __expf(sc[rr] - m); s += sc[rr]; }
        for (int off = 16; off >= 1; off >>= 1) s += __shfl_xor(s, off, 32);
        const float inv = 1.f / s;
#pragma unroll
        for (int rr = 0; rr < 4; ++rr) pbuf[h * 128 + lane * 4 + rr] = sc[rr] * inv;
    }
    __syncthreads();

    const int h2 = t >> 6, j = t & 63;
    float acc = 0.f;
    for (int l = 0; l < 128; ++l)
        acc += pbuf[h2 * 128 + l] *
               bf2f(kv[((size_t)l * 1024 + b) * 512 + 256 + h2 * 64 + j]);
    o[(size_t)b * 256 + h2 * 64 + j] = f2bf(acc);
}

// ------------------------------------------------------------- LayerNorm 256
__global__ void ln256(const float* __restrict__ x,
                      const float* __restrict__ w, const float* __restrict__ bsh,
                      float* __restrict__ outf, unsigned short* __restrict__ outbf,
                      long long M) {
    const int wave = threadIdx.x >> 5, lane = threadIdx.x & 31;
    const long long row = (long long)blockIdx.x * 8 + wave;
    if (row >= M) return;
    const float* xr = x + row * 256;
    float v[8], s = 0.f, s2 = 0.f;
#pragma unroll
    for (int i = 0; i < 8; ++i) {
        v[i] = xr[lane * 8 + i];
        s += v[i]; s2 += v[i] * v[i];
    }
    for (int off = 16; off >= 1; off >>= 1) {
        s  += __shfl_xor(s,  off, 32);
        s2 += __shfl_xor(s2, off, 32);
    }
    const float mu = s * (1.f / 256.f);
    const float var = s2 * (1.f / 256.f) - mu * mu;
    const float rs = rsqrtf(var + 1e-5f);
#pragma unroll
    for (int i = 0; i < 8; ++i) {
        const int c = lane * 8 + i;
        float y = (v[i] - mu) * rs * w[c] + bsh[c];
        if (outf)  outf[row * 256 + c]  = y;
        if (outbf) outbf[row * 256 + c] = f2bf(y);
    }
}

// ------------------------------------------------ group max over l2 (axis=1)
__global__ void max_over_l(const float* __restrict__ ag, float* __restrict__ mx) {
    const int g = blockIdx.x >> 8, q = blockIdx.x & 255, d = threadIdx.x;
    float m = -1e30f;
    for (int l2 = 0; l2 < 128; ++l2) {
        const int p = g * 128 + l2;
        m = fmaxf(m, ag[((size_t)p * 256 + q) * 256 + d]);
    }
    mx[((size_t)g * 256 + q) * 256 + d] = m;
}

// past1[r] = concat(bf16(ag[32768+r]), bf16(mx[i]))     r = i*32768 + l2*256 + q
__global__ void build_past1(const float* __restrict__ ag, const float* __restrict__ mx,
                            unsigned short* __restrict__ past) {
    const long long r = blockIdx.x; const int t = threadIdx.x;
    const int i = (int)(r >> 15), q = (int)(r & 255);
    past[r * 512 + t]       = f2bf(ag[(32768 + r) * 256 + t]);
    past[r * 512 + 256 + t] = f2bf(mx[((size_t)i * 256 + q) * 256 + t]);
}
// past2[r] = concat(bf16(ag[r]), bf16(mx[i+1]))
__global__ void build_past2(const float* __restrict__ ag, const float* __restrict__ mx,
                            unsigned short* __restrict__ past) {
    const long long r = blockIdx.x; const int t = threadIdx.x;
    const int i = (int)(r >> 15), q = (int)(r & 255);
    past[r * 512 + t]       = f2bf(ag[r * 256 + t]);
    past[r * 512 + 256 + t] = f2bf(mx[((size_t)(i + 1) * 256 + q) * 256 + t]);
}

// out rows 1..128 : inter[l2, g*256+q] = (g<3 ? ag2 : ag)[g, l2, q]
__global__ void write_inter(const float* __restrict__ ag, const float* __restrict__ ag2,
                            float* __restrict__ out) {
    const long long blk = blockIdx.x; const int t = threadIdx.x;
    const int g = (int)(blk >> 15), rem = (int)(blk & 32767);
    const int l2 = rem >> 8, q = rem & 255;
    const float v = (g < 3) ? ag2[((size_t)g * 32768 + rem) * 256 + t]
                            : ag[(size_t)(98304 + rem) * 256 + t];
    out[(((size_t)(1 + l2)) * 1024 + g * 256 + q) * 256 + t] = v;
}

// out row 0 + tok output
__global__ void write_token(const float* __restrict__ tok2, float* __restrict__ out) {
    const int bb = blockIdx.x, t = threadIdx.x;
    const float v = tok2[(size_t)bb * 256 + t];
    out[(size_t)bb * 256 + t] = v;
    const int r = bb >> 2, c = bb & 3;
    out[33816576ull + ((size_t)(c * 256 + r)) * 256 + t] = v;
}

// ===========================================================================
extern "C" void kernel_launch(void* const* d_in, const int* in_sizes, int n_in,
                              void* d_out, int out_size, void* d_ws, size_t ws_size,
                              hipStream_t stream) {
    (void)in_sizes; (void)n_in; (void)out_size; (void)ws_size;

    const float* src       = (const float*)d_in[0];
    const float* mix_in_w  = (const float*)d_in[1];  const float* mix_in_b  = (const float*)d_in[2];
    const float* mix_out_w = (const float*)d_in[3];  const float* mix_out_b = (const float*)d_in[4];
    const float* mix_n1_w  = (const float*)d_in[5];  const float* mix_n1_b  = (const float*)d_in[6];
    const float* mix_f1_w  = (const float*)d_in[7];  const float* mix_f1_b  = (const float*)d_in[8];
    const float* mix_f2_w  = (const float*)d_in[9];  const float* mix_f2_b  = (const float*)d_in[10];
    const float* mix_n2_w  = (const float*)d_in[11]; const float* mix_n2_b  = (const float*)d_in[12];
    const float* sa_in_w   = (const float*)d_in[13]; const float* sa_in_b   = (const float*)d_in[14];
    const float* sa_out_w  = (const float*)d_in[15]; const float* sa_out_b  = (const float*)d_in[16];
    const float* l1_w      = (const float*)d_in[17]; const float* l1_b      = (const float*)d_in[18];
    const float* l2_w      = (const float*)d_in[19]; const float* l2_b      = (const float*)d_in[20];
    const float* n1_w      = (const float*)d_in[21]; const float* n1_b      = (const float*)d_in[22];
    const float* n2_w      = (const float*)d_in[23]; const float* n2_b      = (const float*)d_in[24];
    const float* c1_w      = (const float*)d_in[25]; const float* c1_b      = (const float*)d_in[26];
    const float* cn1_w     = (const float*)d_in[27]; const float* cn1_b     = (const float*)d_in[28];
    const float* c2_w      = (const float*)d_in[29]; const float* c2_b      = (const float*)d_in[30];
    const float* cn2_w     = (const float*)d_in[31]; const float* cn2_b     = (const float*)d_in[32];

    float* out = (float*)d_out;
    char* ws = (char*)d_ws;

    const int MB = 131072;           // 128*1024
    // ---- workspace regions (liveness-aliased) -----------------------------
    const size_t R0 = 0;                            // 201326592 B
    const size_t R1 = R0 + 201326592ull;            // 134217728 B
    const size_t R2 = R1 + 134217728ull;
    const size_t R3 = R2 + 134217728ull;
    const size_t R4 = R3 + 134217728ull;
    size_t so = R4 + 134217728ull;                  // smalls
    auto sm = [&](size_t bytes) { size_t o = so; so += (bytes + 255) & ~255ull; return o; };

    unsigned short* w_mix_in  = (unsigned short*)(ws + sm(196608 * 2));
    unsigned short* w_mix_out = (unsigned short*)(ws + sm(65536 * 2));
    unsigned short* w_f1      = (unsigned short*)(ws + sm(131072 * 2));
    unsigned short* w_f2      = (unsigned short*)(ws + sm(131072 * 2));
    unsigned short* w_sa_in   = (unsigned short*)(ws + sm(196608 * 2));
    unsigned short* w_sa_out  = (unsigned short*)(ws + sm(65536 * 2));
    unsigned short* w_l1      = (unsigned short*)(ws + sm(262144 * 2));
    unsigned short* w_l2      = (unsigned short*)(ws + sm(262144 * 2));
    unsigned short* w_c1      = (unsigned short*)(ws + sm(131072 * 2));
    unsigned short* w_c2      = (unsigned short*)(ws + sm(131072 * 2));
    float*          mx1       = (float*)(ws + sm(262144 * 4));
    float*          mx2       = (float*)(ws + sm(262144 * 4));
    unsigned short* q_t       = (unsigned short*)(ws + sm(262144 * 2));
    unsigned short* o_sa      = (unsigned short*)(ws + sm(262144 * 2));
    unsigned short* tok_bf    = (unsigned short*)(ws + sm(262144 * 2));
    float*          tok_pre   = (float*)(ws + sm(262144 * 4));
    float*          tok1      = (float*)(ws + sm(262144 * 4));
    unsigned short* tok1_bf   = (unsigned short*)(ws + sm(262144 * 2));
    unsigned short* ff1_bf    = (unsigned short*)(ws + sm(1048576 * 2));
    float*          t2pre     = (float*)(ws + sm(262144 * 4));
    float*          tok2      = (float*)(ws + sm(262144 * 4));

    unsigned short* qkv_bf  = (unsigned short*)(ws + R0);  // later: h1_bf, past1/2
    unsigned short* h1_bf   = (unsigned short*)(ws + R0);
    unsigned short* past_bf = (unsigned short*)(ws + R0);
    unsigned short* x_bf    = (unsigned short*)(ws + R1);  // later: o_bf, sln_bf, mix_bf, updpre
    unsigned short* o_bf    = (unsigned short*)(ws + R1);
    unsigned short* sln_bf  = (unsigned short*)(ws + R1);
    unsigned short* mix_bf  = (unsigned short*)(ws + R1);
    float*          updpre  = (float*)(ws + R1);
    float*          s_f32   = (float*)(ws + R2);           // later: t_f32, kv_bf
    float*          t_f32   = (float*)(ws + R2);
    unsigned short* kv_bf   = (unsigned short*)(ws + R2);
    float*          sln_f32 = (float*)(ws + R3);           // later: ag2
    float*          ag2     = (float*)(ws + R3);
    float*          ag      = (float*)(ws + R4);           // = mix f32, long-lived

    auto cvt = [&](const float* in, unsigned short* o2, long long n) {
        int blocks = (int)((n + 255) / 256); if (blocks > 131072) blocks = 131072;
        cvt_f32_bf16<<<blocks, 256, 0, stream>>>(in, o2, n);
    };
    auto gemm = [&](const unsigned short* A, const unsigned short* W,
                    const float* bias, const float* res, float* of,
                    unsigned short* ob, int M, int N, int K, int act) {
        dim3 g(N / 128, M / 64);
        gemm_bias<<<g, 256, 0, stream>>>(A, W, bias, res, of, ob, M, N, K, act);
    };

    // 1) weights -> bf16
    cvt(mix_in_w,  w_mix_in,  196608);  cvt(mix_out_w, w_mix_out, 65536);
    cvt(mix_f1_w,  w_f1,      131072);  cvt(mix_f2_w,  w_f2,      131072);
    cvt(sa_in_w,   w_sa_in,   196608);  cvt(sa_out_w,  w_sa_out,  65536);
    cvt(l1_w,      w_l1,      262144);  cvt(l2_w,      w_l2,      262144);
    cvt(c1_w,      w_c1,      131072);  cvt(c2_w,      w_c2,      131072);

    // 2) x = src[1:] -> bf16
    cvt(src + 262144, x_bf, (long long)MB * 256);

    // 3) qkv = x @ mix_in_w^T + b
    gemm(x_bf, w_mix_in, mix_in_b, nullptr, nullptr, qkv_bf, MB, 768, 256, 0);
    // 4) mix self-attention
    attn_mix<<<8192, 256, 0, stream>>>(qkv_bf, o_bf);
    // 5) s = x + o @ mix_out_w^T + b
    gemm(o_bf, w_mix_out, mix_out_b, src + 262144, s_f32, nullptr, MB, 256, 256, 0);
    // 6) s = LN(s)
    ln256<<<MB / 8, 256, 0, stream>>>(s_f32, mix_n1_w, mix_n1_b, sln_f32, sln_bf, MB);
    // 7/8) FFN + residual
    gemm(sln_bf, w_f1, mix_f1_b, nullptr, nullptr, h1_bf, MB, 512, 256, 1);
    gemm(h1_bf, w_f2, mix_f2_b, sln_f32, t_f32, nullptr, MB, 256, 512, 0);
    // 9) mix = LN(...)   (ag == mix, flat layouts identical)
    ln256<<<MB / 8, 256, 0, stream>>>(t_f32, mix_n2_w, mix_n2_b, ag, mix_bf, MB);

    // ---- token branch
    cvt(src, tok_bf, 262144);
    gemm(tok_bf, w_sa_in, sa_in_b, nullptr, nullptr, q_t, 1024, 256, 256, 0);
    gemm(mix_bf, w_sa_in + (size_t)256 * 256, sa_in_b + 256, nullptr, nullptr,
         kv_bf, MB, 512, 256, 0);
    attn_sa<<<1024, 256, 0, stream>>>(q_t, kv_bf, o_sa);
    gemm(o_sa, w_sa_out, sa_out_b, src, tok_pre, nullptr, 1024, 256, 256, 0);
    ln256<<<128, 256, 0, stream>>>(tok_pre, n1_w, n1_b, tok1, tok1_bf, 1024);
    gemm(tok1_bf, w_l1, l1_b, nullptr, nullptr, ff1_bf, 1024, 1024, 256, 1);
    gemm(ff1_bf, w_l2, l2_b, tok1, t2pre, nullptr, 1024, 256, 1024, 0);
    ln256<<<128, 256, 0, stream>>>(t2pre, n2_w, n2_b, tok2, nullptr, 1024);

    // ---- grouped stage 1: upd = LN(past1 @ c1^T + c1_b + ag[1:])
    max_over_l<<<1024, 256, 0, stream>>>(ag, mx1);
    build_past1<<<98304, 256, 0, stream>>>(ag, mx1, past_bf);
    gemm(past_bf, w_c1, c1_b, ag + (size_t)32768 * 256, updpre, nullptr,
         98304, 256, 512, 0);
    ln256<<<98304 / 8, 256, 0, stream>>>(updpre, cn1_w, cn1_b,
                                         ag + (size_t)32768 * 256, nullptr, 98304);
    // ---- grouped stage 2: upd2 = LN(past2 @ c2^T + c2_b + ag[:-1])
    max_over_l<<<1024, 256, 0, stream>>>(ag, mx2);
    build_past2<<<98304, 256, 0, stream>>>(ag, mx2, past_bf);
    gemm(past_bf, w_c2, c2_b, ag, updpre, nullptr, 98304, 256, 512, 0);
    ln256<<<98304 / 8, 256, 0, stream>>>(updpre, cn2_w, cn2_b, ag2, nullptr, 98304);

    // ---- assemble outputs
    write_inter<<<131072, 256, 0, stream>>>(ag, ag2, out);
    write_token<<<1024, 256, 0, stream>>>(tok2, out);
}
